// TrajLSTM_8744553415029
// MI455X (gfx1250) — compile-verified, matrix operand order checked by reading, not run on hardware
//
#include <hip/hip_runtime.h>

#define B_  1024
#define T_  64
#define I_  64
#define H_  1024
#define F_  16

typedef __bf16 bf16;
typedef __attribute__((ext_vector_type(16))) __bf16 v16bf;
typedef __attribute__((ext_vector_type(8)))  __bf16 v8bf;
typedef __attribute__((ext_vector_type(8)))  float  v8f;

// WMMA 16-bit A-layout fragment from an LDS tile with leading dim 32.
// Lane l holds row M=l%16, K-chunks [half*8..+7] and [16+half*8..+7].
__device__ __forceinline__ v16bf lds_frag(const bf16* s, int row_base) {
  const int lane = threadIdx.x & 31;
  const int m    = lane & 15;
  const int half = lane >> 4;
  const bf16* p  = s + (row_base + m) * 32 + half * 8;
  v8bf lo = *(const v8bf*)p;
  v8bf hi = *(const v8bf*)(p + 16);
  v16bf r;
#pragma unroll
  for (int i = 0; i < 8; ++i) { r[i] = lo[i]; r[i + 8] = hi[i]; }
  return r;
}

// Direct-from-global fragment (used by the small linear kernel).
__device__ __forceinline__ v16bf load_frag(const bf16* __restrict__ base, long ld,
                                           int row_base, int col_base) {
  const int lane = threadIdx.x & 31;
  const int m    = lane & 15;
  const int half = lane >> 4;
  const bf16* p  = base + (long)(row_base + m) * ld + col_base + half * 8;
  v8bf lo = *(const v8bf*)(p);
  v8bf hi = *(const v8bf*)(p + 16);
  v16bf r;
#pragma unroll
  for (int i = 0; i < 8; ++i) { r[i] = lo[i]; r[i + 8] = hi[i]; }
  return r;
}

__device__ __forceinline__ float sigmoidf_(float x) {
  return 1.0f / (1.0f + __expf(-x));
}
__device__ __forceinline__ float tanhf_(float x) {
  x = fminf(fmaxf(x, -15.0f), 15.0f);
  float e = __expf(2.0f * x);
  return (e - 1.0f) / (e + 1.0f);
}

// Fused LSTM layer step with LDS-staged operands:
//   gates = [A1, Hprev] @ [Wx | Wh]^T + bih + bhh, then cell update.
// Workgroup tile: 64 batch rows x 64 h-cols (x4 gates => GEMM N=256).
// Per 32-K chunk: A tile 64x32 (4KB) + W tile 256x32 (16KB) staged once;
// all 8 waves read fragments via ds_load_b128. Double-buffered, one
// barrier per chunk, manual 2x unroll so buffer offsets are immediates.
__global__ __launch_bounds__(256) void lstm_gates_kernel(
    const bf16* __restrict__ A1, long ldA1, int K1,
    const bf16* __restrict__ Hprev,
    const bf16* __restrict__ Wx, const bf16* __restrict__ Wh,
    const float* __restrict__ bih, const float* __restrict__ bhh,
    float* __restrict__ C, bf16* __restrict__ Hout)
{
  __shared__ bf16 As[2][64 * 32];    //  2 x 4KB
  __shared__ bf16 Bs[2][256 * 32];   //  2 x 16KB

  const int tid  = threadIdx.x;
  const int wave = tid >> 5;
  const int wrow = wave >> 2;                 // 0..1
  const int wcol = wave & 3;                  // 0..3
  const int mblk = blockIdx.x * 64;
  const int nblk = blockIdx.y * 64;

  const int nc = (K1 + H_) / 32;              // chunks never straddle K1 boundary

  // Staging assignment: thread t owns A vec (row=t>>2, colvec=t&3) and the
  // same (row,colvec) of each of the 4 gate sub-tiles of W.
  const int arow = tid >> 2;                  // 0..63
  const int acv  = (tid & 3) * 8;             // 0,8,16,24 (bf16 elems)

  v8bf areg;
  v8bf breg[4];

  auto gfetch = [&](int c) {
    const int kglob = c * 32;
    const bf16* ap; long la; int ko;
    const bf16* wp; long lw;
    if (kglob < K1) { ap = A1;    la = ldA1; ko = kglob;      wp = Wx; lw = K1; }
    else            { ap = Hprev; la = H_;   ko = kglob - K1; wp = Wh; lw = H_; }
    areg = *(const v8bf*)(ap + (long)(mblk + arow) * la + ko + acv);
#pragma unroll
    for (int g = 0; g < 4; ++g)
      breg[g] = *(const v8bf*)(wp + (long)(g * H_ + nblk + arow) * lw + ko + acv);
  };

  v8f acc[2][4];
#pragma unroll
  for (int mt = 0; mt < 2; ++mt)
#pragma unroll
    for (int g = 0; g < 4; ++g)
#pragma unroll
      for (int e = 0; e < 8; ++e) acc[mt][g][e] = 0.0f;

  // One 32-K chunk: issue all 12 ds_load_b128 (6 fragments in distinct
  // registers) before the 8 WMMAs so the compiler can use partial
  // s_wait_dscnt and keep LDS loads in flight behind the matrix ops.
#define CHUNK(BUF)                                                              \
  do {                                                                          \
    *(v8bf*)(&As[BUF][arow * 32 + acv]) = areg;                                 \
    _Pragma("unroll")                                                           \
    for (int g = 0; g < 4; ++g)                                                 \
      *(v8bf*)(&Bs[BUF][(g * 64 + arow) * 32 + acv]) = breg[g];                 \
    if (c + 1 < nc) gfetch(c + 1);                                              \
    __syncthreads();                                                            \
    v16bf a0 = lds_frag(As[BUF], wrow * 32);                                    \
    v16bf a1 = lds_frag(As[BUF], wrow * 32 + 16);                               \
    v16bf w0 = lds_frag(Bs[BUF], 0 * 64 + wcol * 16);                           \
    v16bf w1 = lds_frag(Bs[BUF], 1 * 64 + wcol * 16);                           \
    v16bf w2 = lds_frag(Bs[BUF], 2 * 64 + wcol * 16);                           \
    v16bf w3 = lds_frag(Bs[BUF], 3 * 64 + wcol * 16);                           \
    acc[0][0] = __builtin_amdgcn_wmma_f32_16x16x32_bf16(false, a0, false, w0,   \
                                      (short)0, acc[0][0], false, false);       \
    acc[1][0] = __builtin_amdgcn_wmma_f32_16x16x32_bf16(false, a1, false, w0,   \
                                      (short)0, acc[1][0], false, false);       \
    acc[0][1] = __builtin_amdgcn_wmma_f32_16x16x32_bf16(false, a0, false, w1,   \
                                      (short)0, acc[0][1], false, false);       \
    acc[1][1] = __builtin_amdgcn_wmma_f32_16x16x32_bf16(false, a1, false, w1,   \
                                      (short)0, acc[1][1], false, false);       \
    acc[0][2] = __builtin_amdgcn_wmma_f32_16x16x32_bf16(false, a0, false, w2,   \
                                      (short)0, acc[0][2], false, false);       \
    acc[1][2] = __builtin_amdgcn_wmma_f32_16x16x32_bf16(false, a1, false, w2,   \
                                      (short)0, acc[1][2], false, false);       \
    acc[0][3] = __builtin_amdgcn_wmma_f32_16x16x32_bf16(false, a0, false, w3,   \
                                      (short)0, acc[0][3], false, false);       \
    acc[1][3] = __builtin_amdgcn_wmma_f32_16x16x32_bf16(false, a1, false, w3,   \
                                      (short)0, acc[1][3], false, false);       \
  } while (0)

  gfetch(0);
  {
    int c = 0;
    for (;;) {
      CHUNK(0);                 // buffer 0: compile-time LDS offsets
      if (++c == nc) break;
      CHUNK(1);                 // buffer 1
      if (++c == nc) break;
    }
  }
#undef CHUNK

  // Elementwise cell update. D layout: VGPR r -> M = r + 8*(lane>=16), N = lane%16.
  const int lane   = threadIdx.x & 31;
  const int n      = nblk + wcol * 16 + (lane & 15);
  const int rowoff = (lane >> 4) * 8;
  const int mbase  = mblk + wrow * 32;
  const float bi = bih[0 * H_ + n] + bhh[0 * H_ + n];
  const float bf = bih[1 * H_ + n] + bhh[1 * H_ + n];
  const float bg = bih[2 * H_ + n] + bhh[2 * H_ + n];
  const float bo = bih[3 * H_ + n] + bhh[3 * H_ + n];

#pragma unroll
  for (int mt = 0; mt < 2; ++mt) {
#pragma unroll
    for (int r = 0; r < 8; ++r) {
      const int  brow = mbase + mt * 16 + rowoff + r;
      const long idx  = (long)brow * H_ + n;
      const float ii = sigmoidf_(acc[mt][0][r] + bi);
      const float ff = sigmoidf_(acc[mt][1][r] + bf);
      const float gg = tanhf_  (acc[mt][2][r] + bg);
      const float oo = sigmoidf_(acc[mt][3][r] + bo);
      const float cn = ff * C[idx] + ii * gg;
      C[idx]    = cn;
      Hout[idx] = (bf16)(oo * tanhf_(cn));
    }
  }
}

// out = H2 @ Wl^T + bl  (B x H) x (I x H)^T -> (B x I); also emit bf16 copy
// for the autoregressive feedback path. One wave per block, 16 rows x 64 cols.
__global__ __launch_bounds__(32) void linear_kernel(
    const bf16* __restrict__ H2, const bf16* __restrict__ Wl,
    const float* __restrict__ bl,
    float* __restrict__ out, long out_row_stride,
    bf16* __restrict__ xnext)
{
  const int mbase = blockIdx.x * 16;
  v8f acc[4];
#pragma unroll
  for (int j = 0; j < 4; ++j)
#pragma unroll
    for (int e = 0; e < 8; ++e) acc[j][e] = 0.0f;

  for (int k = 0; k < H_; k += 32) {
    v16bf a = load_frag(H2, H_, mbase, k);
#pragma unroll
    for (int j = 0; j < 4; ++j) {
      v16bf wb = load_frag(Wl, H_, j * 16, k);
      acc[j] = __builtin_amdgcn_wmma_f32_16x16x32_bf16(false, a, false, wb,
                                                       (short)0, acc[j], false, false);
    }
  }

  const int lane   = threadIdx.x & 31;
  const int nin    = lane & 15;
  const int rowoff = (lane >> 4) * 8;
#pragma unroll
  for (int j = 0; j < 4; ++j) {
    const int col  = j * 16 + nin;
    const float bb = bl[col];
#pragma unroll
    for (int r = 0; r < 8; ++r) {
      const int row = mbase + rowoff + r;
      const float v = acc[j][r] + bb;
      out[(long)row * out_row_stride + col] = v;
      xnext[row * I_ + col] = (bf16)v;
    }
  }
}

__global__ void f32_to_bf16_kernel(const float* __restrict__ s, bf16* __restrict__ d, int n) {
  int i = blockIdx.x * blockDim.x + threadIdx.x;
  if (i < n) d[i] = (bf16)s[i];
}

__global__ void zero_f32_kernel(float* __restrict__ p, int n) {
  int i = blockIdx.x * blockDim.x + threadIdx.x;
  if (i < n) p[i] = 0.0f;
}

extern "C" void kernel_launch(void* const* d_in, const int* in_sizes, int n_in,
                              void* d_out, int out_size, void* d_ws, size_t ws_size,
                              hipStream_t stream) {
  const float* x    = (const float*)d_in[0];
  const float* Wih1 = (const float*)d_in[1];
  const float* Whh1 = (const float*)d_in[2];
  const float* bih1 = (const float*)d_in[3];
  const float* bhh1 = (const float*)d_in[4];
  const float* Wih2 = (const float*)d_in[5];
  const float* Whh2 = (const float*)d_in[6];
  const float* bih2 = (const float*)d_in[7];
  const float* bhh2 = (const float*)d_in[8];
  const float* Wlin = (const float*)d_in[9];
  const float* blin = (const float*)d_in[10];
  float* out = (float*)d_out;

  char*  wsp = (char*)d_ws;
  size_t off = 0;
  auto carve = [&](size_t bytes) -> char* {
    char* p = wsp + off;
    off = (off + bytes + 255) & ~(size_t)255;
    return p;
  };
  bf16* x_bf = (bf16*)carve((size_t)B_ * T_ * I_ * 2);
  bf16* w1x  = (bf16*)carve((size_t)4 * H_ * I_ * 2);
  bf16* w1h  = (bf16*)carve((size_t)4 * H_ * H_ * 2);
  bf16* w2x  = (bf16*)carve((size_t)4 * H_ * H_ * 2);
  bf16* w2h  = (bf16*)carve((size_t)4 * H_ * H_ * 2);
  bf16* wl   = (bf16*)carve((size_t)I_ * H_ * 2);
  bf16* h1[2] = { (bf16*)carve((size_t)B_ * H_ * 2), (bf16*)carve((size_t)B_ * H_ * 2) };
  bf16* h2[2] = { (bf16*)carve((size_t)B_ * H_ * 2), (bf16*)carve((size_t)B_ * H_ * 2) };
  float* c1  = (float*)carve((size_t)B_ * H_ * 4);
  float* c2  = (float*)carve((size_t)B_ * H_ * 4);
  bf16* xnext = (bf16*)carve((size_t)B_ * I_ * 2);

  auto cvt = [&](const float* s, bf16* d, int n) {
    f32_to_bf16_kernel<<<(n + 255) / 256, 256, 0, stream>>>(s, d, n);
  };
  cvt(x,    x_bf, B_ * T_ * I_);
  cvt(Wih1, w1x,  4 * H_ * I_);
  cvt(Whh1, w1h,  4 * H_ * H_);
  cvt(Wih2, w2x,  4 * H_ * H_);
  cvt(Whh2, w2h,  4 * H_ * H_);
  cvt(Wlin, wl,   I_ * H_);

  auto zf = [&](float* p, int n) {
    zero_f32_kernel<<<(n + 255) / 256, 256, 0, stream>>>(p, n);
  };
  zf((float*)h1[0], B_ * H_ / 2);
  zf((float*)h1[1], B_ * H_ / 2);
  zf((float*)h2[0], B_ * H_ / 2);
  zf((float*)h2[1], B_ * H_ / 2);
  zf(c1, B_ * H_);
  zf(c2, B_ * H_);

  const int NSTEP = T_ + F_;
  const dim3 ggrid(B_ / 64, H_ / 64);
  for (int t = 0; t < NSTEP; ++t) {
    const bf16* a1;
    long lda1;
    if (t < T_) { a1 = x_bf + (size_t)t * I_; lda1 = (long)T_ * I_; }
    else        { a1 = xnext;                 lda1 = I_; }
    const int pi = t & 1, po = pi ^ 1;
    lstm_gates_kernel<<<ggrid, 256, 0, stream>>>(a1, lda1, I_, h1[pi],
                                                 w1x, w1h, bih1, bhh1, c1, h1[po]);
    lstm_gates_kernel<<<ggrid, 256, 0, stream>>>(h1[po], (long)H_, H_, h2[pi],
                                                 w2x, w2h, bih2, bhh2, c2, h2[po]);
    linear_kernel<<<B_ / 16, 32, 0, stream>>>(h2[po], wl, blin,
                                              out + (size_t)t * I_,
                                              (long)NSTEP * I_, xnext);
  }
  (void)in_sizes; (void)n_in; (void)out_size; (void)ws_size;
}
